// FastSpeech_11879879542878
// MI455X (gfx1250) — matrix operand order repaired and checked by reading, not compiled, not run on hardware
//
#include <hip/hip_runtime.h>

// ---------------------------------------------------------------------------
// FastSpeech inference for MI455X (gfx1250).
// All GEMMs via v_wmma_f32_16x16x32_f16 (f32 staged in LDS, converted to f16
// with v_cvt_pk_rtz_f16_f32 during fragment build; f32 accumulate).
// Stage-in: double-buffered global_load_async_to_lds_b128 (ASYNCcnt-pipelined,
// next K-tile DMA overlaps current tile's WMMAs); sync b128 copy fallback.
// ~142 MB workspace.
// ---------------------------------------------------------------------------

typedef __attribute__((ext_vector_type(16))) _Float16 v16h;
typedef __attribute__((ext_vector_type(2)))  __fp16   hf2;
typedef __attribute__((ext_vector_type(8)))  float    v8f;
typedef __attribute__((ext_vector_type(4)))  int      v4i;

union FragU { v16h h; unsigned int u[8]; };

constexpr int Bc = 16, Lc = 128, Tc = 1024;
constexpr int Dc = 384, Hc = 2, Fc = 1536, NLc = 6;
constexpr int NMELc = 80, CDPc = 256;
constexpr int DHc = Dc / Hc;          // 192

// GEMM tiling
constexpr int GBM = 64, GBN = 128, GBK = 32;
constexpr int PIT_A = 36;    // f32 words/row for A [row][k]  (16B aligned, conflict-free)
constexpr int PIT_BN = 36;   // f32 words/row for B [n][k]    (TB path)
constexpr int PIT_BK = 132;  // f32 words/row for B [k][n]    (NN path)

// Optional CDNA5 async global->LDS staging (ASYNCcnt-tracked)
#if defined(__has_builtin)
#if __has_builtin(__builtin_amdgcn_global_load_async_to_lds_b128) && \
    __has_builtin(__builtin_amdgcn_s_wait_asynccnt)
#define USE_ASYNC_LDS 1
#endif
#endif
#ifndef USE_ASYNC_LDS
#define USE_ASYNC_LDS 0
#endif

__device__ __forceinline__ void copy16(const float* __restrict__ g, float* __restrict__ l)
{
#if USE_ASYNC_LDS
  __builtin_amdgcn_global_load_async_to_lds_b128(
      (__attribute__((address_space(1))) v4i*)g,
      (__attribute__((address_space(3))) v4i*)l, 0, 0);
#else
  *(float4*)l = *(const float4*)g;
#endif
}

__device__ __forceinline__ unsigned pkcvt(float a, float b)
{
  union { hf2 h; unsigned u; } c;
  c.h = __builtin_amdgcn_cvt_pkrtz(a, b);   // v_cvt_pk_rtz_f16_f32
  return c.u;
}

// ---------------------------------------------------------------------------
// WMMA GEMM:  C[M,N] = alpha * (A[M,K] @ op(B)) + bias[N]   (optional ReLU)
//   TB=false: B is [K,N] row-major.  TB=true: B is [N,K] row-major (A@B^T).
// grid.z batches via element strides bsA/bsB/bsC.
// Preconditions (all call sites): M % 64 == 0, K % 32 == 0, N % 4 == 0,
// and for TB=true additionally N % 128 == 0.
// Each wave stages exactly 6 16B async chunks per K-tile (2 A + 4 B), so
// s_wait_asynccnt 6 retires the current tile while the prefetch stays in
// flight under the WMMAs (double-buffered LDS ping-pong).
// ---------------------------------------------------------------------------
template <bool TB, bool RELU>
__global__ __launch_bounds__(256) void gemm_wmma(
    const float* __restrict__ A, int lda, size_t bsA,
    const float* __restrict__ Bm, int ldb, size_t bsB,
    float* __restrict__ C, int ldc, size_t bsC,
    const float* __restrict__ bias,
    int M, int N, int K, float alpha)
{
  (void)M;
  A  += (size_t)blockIdx.z * bsA + (size_t)(blockIdx.y * GBM) * lda;
  Bm += (size_t)blockIdx.z * bsB;
  C  += (size_t)blockIdx.z * bsC;

  const int n0 = blockIdx.x * GBN;
  const int m0 = blockIdx.y * GBM;
  const int tid  = threadIdx.x;
  const int wave = tid >> 5;
  const int lane = tid & 31;
  const int wm = wave >> 2;          // 0..1 -> 32 rows
  const int wn = wave & 3;           // 0..3 -> 32 cols
  const int hl = lane & 15;
  const int hi = lane >> 4;

  __shared__ float sA[2][GBM * PIT_A];    // 2 x  9216 B
  __shared__ float sB[2][GBN * PIT_BN];   // 2 x 18432 B (>= GBK*PIT_BK)

  auto stage = [&](int k0, int buf) {
    float* dA = sA[buf];
    float* dB = sB[buf];
    // A tile 64x32 f32: 512 16B-chunks, 2 per thread
#pragma unroll
    for (int i = 0; i < 2; i++) {
      int id = tid + i * 256;
      int r = id >> 3, c4 = (id & 7) * 4;
      copy16(A + (size_t)r * lda + (k0 + c4), &dA[r * PIT_A + c4]);
    }
    // B tile: 1024 16B-chunks, 4 per thread
    if constexpr (TB) {
      // B[N,K] row-major -> LDS [n][k]
#pragma unroll
      for (int i = 0; i < 4; i++) {
        int id = tid + i * 256;
        int r = id >> 3, c4 = (id & 7) * 4;
        copy16(Bm + (size_t)(n0 + r) * ldb + (k0 + c4), &dB[r * PIT_BN + c4]);
      }
    } else {
      // B[K,N] row-major -> LDS [k][n] (chunk-clamped N guard; garbage cols
      // are finite and discarded by the guarded store)
#pragma unroll
      for (int i = 0; i < 4; i++) {
        int id = tid + i * 256;
        int kk = id >> 5, n4 = (id & 31) * 4;
        int gn = n0 + n4;
        if (gn > N - 4) gn = N - 4;
        copy16(Bm + (size_t)(k0 + kk) * ldb + gn, &dB[kk * PIT_BK + n4]);
      }
    }
  };

  v8f zero = {0.f, 0.f, 0.f, 0.f, 0.f, 0.f, 0.f, 0.f};
  v8f acc[2][2] = {{zero, zero}, {zero, zero}};

  stage(0, 0);                     // prologue: issue first tile
  int cur = 0;
  for (int k0 = 0; k0 < K; k0 += GBK, cur ^= 1) {
    if (k0 + GBK < K) {
      stage(k0 + GBK, cur ^ 1);    // prefetch next tile into other buffer
#if USE_ASYNC_LDS
      __builtin_amdgcn_s_wait_asynccnt(6);   // current tile landed; prefetch in flight
#endif
    } else {
#if USE_ASYNC_LDS
      __builtin_amdgcn_s_wait_asynccnt(0);
#endif
    }
    __syncthreads();               // all waves' current-tile data visible

    // ---- build f16 fragments from f32 LDS (pack pairs with cvt_pk_rtz) ----
    FragU afr[2], bfr[2];
#pragma unroll
    for (int ti = 0; ti < 2; ti++) {
      const float* ap = &sA[cur][(wm * 32 + ti * 16 + hl) * PIT_A];
#pragma unroll
      for (int j = 0; j < 8; j++) {
        int kk = hi * 16 + 2 * j;
        float2 p = *(const float2*)(ap + kk);
        afr[ti].u[j] = pkcvt(p.x, p.y);
      }
    }
#pragma unroll
    for (int tj = 0; tj < 2; tj++) {
      int col = wn * 32 + tj * 16 + hl;
      int kb = hi * 16;
      if constexpr (TB) {
        const float* bp = &sB[cur][col * PIT_BN + kb];
#pragma unroll
        for (int j = 0; j < 8; j++) {
          float2 p = *(const float2*)(bp + 2 * j);
          bfr[tj].u[j] = pkcvt(p.x, p.y);
        }
      } else {
#pragma unroll
        for (int j = 0; j < 8; j++) {
          float a = sB[cur][(kb + 2 * j) * PIT_BK + col];
          float b = sB[cur][(kb + 2 * j + 1) * PIT_BK + col];
          bfr[tj].u[j] = pkcvt(a, b);
        }
      }
    }

#pragma unroll
    for (int ti = 0; ti < 2; ti++)
#pragma unroll
      for (int tj = 0; tj < 2; tj++)
        acc[ti][tj] = __builtin_amdgcn_wmma_f32_16x16x32_f16(
            false, afr[ti].h, false, bfr[tj].h,
            (short)0, acc[ti][tj], false, false);

    __syncthreads();               // reads done before next iter overwrites 'cur'
  }

  // ---- store (C layout: lane -> col lane&15, VGPR r -> row r + 8*hi) ----
#pragma unroll
  for (int ti = 0; ti < 2; ti++)
#pragma unroll
    for (int tj = 0; tj < 2; tj++) {
      int col = n0 + wn * 32 + tj * 16 + hl;
      if (col < N) {
        float bv = bias ? bias[col] : 0.f;
#pragma unroll
        for (int r = 0; r < 8; r++) {
          int row = m0 + wm * 32 + ti * 16 + hi * 8 + r;
          float v = acc[ti][tj][r] * alpha + bv;
          if (RELU) v = fmaxf(v, 0.f);
          C[(size_t)row * ldc + col] = v;
        }
      }
    }
}

// ---------------------------------------------------------------------------
// Elementwise / reduction kernels
// ---------------------------------------------------------------------------
__device__ __forceinline__ float posenc(int pos, int d)
{
  float fi = (float)(d & ~1);
  float ang = (float)pos * __expf(fi * (-9.210340371976184f / (float)Dc)); // ln(1e4)
  return (d & 1) ? __cosf(ang) : __sinf(ang);
}

__global__ void embed_pos_kernel(const int* __restrict__ ph,
                                 const float* __restrict__ emb,
                                 const float* __restrict__ alpha,
                                 float* __restrict__ x, int n)
{
  int i = blockIdx.x * 256 + threadIdx.x;
  if (i >= n) return;
  int d = i % Dc;
  int bl = i / Dc;
  int l = bl % Lc;
  x[i] = emb[(size_t)ph[bl] * Dc + d] + alpha[0] * posenc(l, d);
}

__global__ void mask_enc_kernel(const int* __restrict__ ph, float* __restrict__ m, int n)
{
  int i = blockIdx.x * 256 + threadIdx.x;
  if (i < n) m[i] = (ph[i] == 0) ? 1.f : 0.f;
}

// out[row] = LN(a[row] (+ r[row]))
__global__ __launch_bounds__(128) void ln_kernel(
    const float* __restrict__ a, const float* __restrict__ r,
    const float* __restrict__ gamma, const float* __restrict__ beta,
    float* __restrict__ out, int C)
{
  int row = blockIdx.x, t = threadIdx.x;
  const float* ap = a + (size_t)row * C;
  const float* rp = r ? (r + (size_t)row * C) : nullptr;
  float v[4];
  float s = 0.f;
  int j = 0;
  for (int i = t; i < C; i += 128, j++) {
    float x = ap[i] + (rp ? rp[i] : 0.f);
    v[j] = x; s += x;
  }
  __shared__ float red[128];
  red[t] = s; __syncthreads();
  for (int o = 64; o > 0; o >>= 1) { if (t < o) red[t] += red[t + o]; __syncthreads(); }
  float mean = red[0] / (float)C;
  __syncthreads();
  float s2 = 0.f;
  j = 0;
  for (int i = t; i < C; i += 128, j++) { float d = v[j] - mean; s2 += d * d; }
  red[t] = s2; __syncthreads();
  for (int o = 64; o > 0; o >>= 1) { if (t < o) red[t] += red[t + o]; __syncthreads(); }
  float inv = rsqrtf(red[0] / (float)C + 1e-6f);
  j = 0;
  for (int i = t; i < C; i += 128, j++)
    out[(size_t)row * C + i] = (v[j] - mean) * inv * gamma[i] + beta[i];
}

// masked softmax over last dim of logits[H,S,S]; maskf[k]==1 -> masked
__global__ __launch_bounds__(256) void softmax_kernel(
    float* __restrict__ logits, const float* __restrict__ maskf, int S)
{
  int q = blockIdx.x, h = blockIdx.y, t = threadIdx.x;
  float* row = logits + ((size_t)h * S + q) * S;
  __shared__ float red[256];
  float lm = -3.0e38f;
  for (int i = t; i < S; i += 256) {
    float v = row[i] - maskf[i] * 1e9f;
    lm = fmaxf(lm, v);
  }
  red[t] = lm; __syncthreads();
  for (int o = 128; o > 0; o >>= 1) { if (t < o) red[t] = fmaxf(red[t], red[t + o]); __syncthreads(); }
  float m = red[0];
  __syncthreads();
  float ls = 0.f;
  for (int i = t; i < S; i += 256) {
    float v = __expf(row[i] - maskf[i] * 1e9f - m);
    row[i] = v; ls += v;
  }
  red[t] = ls; __syncthreads();
  for (int o = 128; o > 0; o >>= 1) { if (t < o) red[t] += red[t + o]; __syncthreads(); }
  float inv = 1.0f / red[0];
  for (int i = t; i < S; i += 256) row[i] *= inv;
}

// K=3 SAME conv + bias + relu
__global__ __launch_bounds__(256) void conv3_relu_kernel(
    const float* __restrict__ x, const float* __restrict__ w,
    const float* __restrict__ bias, float* __restrict__ out,
    int S, int Cin, int Cout)
{
  int l = blockIdx.x, b = blockIdx.y, t = threadIdx.x;
  __shared__ float sx[3 * Dc];
  for (int i = t; i < 3 * Cin; i += blockDim.x) {
    int kt = i / Cin, ci = i - kt * Cin;
    int ll = l + kt - 1;
    sx[i] = (ll >= 0 && ll < S) ? x[((size_t)b * S + ll) * Cin + ci] : 0.f;
  }
  __syncthreads();
  if (t < Cout) {
    float acc = bias[t];
    for (int kt = 0; kt < 3; kt++)
      for (int ci = 0; ci < Cin; ci++)
        acc += sx[kt * Cin + ci] * w[((size_t)kt * Cin + ci) * Cout + t];
    out[((size_t)b * S + l) * Cout + t] = fmaxf(acc, 0.f);
  }
}

__global__ __launch_bounds__(256) void dp_dot_kernel(
    const float* __restrict__ x, const float* __restrict__ w,
    const float* __restrict__ b0, float* __restrict__ out)
{
  int row = blockIdx.x, t = threadIdx.x;
  __shared__ float red[CDPc];
  red[t] = x[(size_t)row * CDPc + t] * w[t];
  __syncthreads();
  for (int o = CDPc / 2; o > 0; o >>= 1) { if (t < o) red[t] += red[t + o]; __syncthreads(); }
  if (t == 0) out[row] = red[0] + b0[0];
}

__global__ __launch_bounds__(128) void cumsum_kernel(const int* __restrict__ dur, int* __restrict__ cum)
{
  int b = blockIdx.x, t = threadIdx.x;
  __shared__ int s[Lc];
  s[t] = dur[b * Lc + t];
  __syncthreads();
  for (int o = 1; o < Lc; o <<= 1) {
    int v = (t >= o) ? s[t - o] : 0;
    __syncthreads();
    s[t] += v;
    __syncthreads();
  }
  cum[b * Lc + t] = s[t];
}

__global__ void lr_idx_kernel(const int* __restrict__ cum, int* __restrict__ idx,
                              float* __restrict__ maskd, int n)
{
  int i = blockIdx.x * 256 + threadIdx.x;
  if (i >= n) return;
  int t = i % Tc, b = i / Tc;
  const int* c = cum + b * Lc;
  int lo = 0, hi = Lc;
  while (lo < hi) {                      // searchsorted(c, t, 'right')
    int mid = (lo + hi) >> 1;
    if (c[mid] <= t) lo = mid + 1; else hi = mid;
  }
  idx[i] = (lo < Lc) ? lo : (Lc - 1);
  maskd[i] = (t < c[Lc - 1]) ? 0.f : 1.f;
}

__global__ void gather_pos_kernel(const float* __restrict__ henc, const int* __restrict__ idx,
                                  const float* __restrict__ maskd, const float* __restrict__ alpha,
                                  float* __restrict__ y, int n)
{
  int i = blockIdx.x * 256 + threadIdx.x;
  if (i >= n) return;
  int d = i % Dc;
  int bt = i / Dc;
  int t = bt % Tc, b = bt / Tc;
  float v = (maskd[bt] == 0.f) ? henc[((size_t)b * Lc + idx[bt]) * Dc + d] : 0.f;
  y[i] = v + alpha[0] * posenc(t, d);
}

// ---------------------------------------------------------------------------
// Host orchestration
// ---------------------------------------------------------------------------
static inline void launch_gemm(hipStream_t st,
                               const float* A, int lda, size_t bsA,
                               const float* Bm, int ldb, size_t bsB,
                               float* C, int ldc, size_t bsC,
                               const float* bias, int M, int N, int K,
                               float alpha, int batch, bool tb, bool relu)
{
  dim3 g((N + GBN - 1) / GBN, (M + GBM - 1) / GBM, batch), blk(256);
  if (tb) {
    if (relu) gemm_wmma<true, true><<<g, blk, 0, st>>>(A, lda, bsA, Bm, ldb, bsB, C, ldc, bsC, bias, M, N, K, alpha);
    else      gemm_wmma<true, false><<<g, blk, 0, st>>>(A, lda, bsA, Bm, ldb, bsB, C, ldc, bsC, bias, M, N, K, alpha);
  } else {
    if (relu) gemm_wmma<false, true><<<g, blk, 0, st>>>(A, lda, bsA, Bm, ldb, bsB, C, ldc, bsC, bias, M, N, K, alpha);
    else      gemm_wmma<false, false><<<g, blk, 0, st>>>(A, lda, bsA, Bm, ldb, bsB, C, ldc, bsC, bias, M, N, K, alpha);
  }
}

static void run_stack(hipStream_t st, int S,
                      float* X, float* Q, float* Kb, float* Vb, float* Tb, float* LOG,
                      const float* maskf,
                      const float* aw, const float* ab,
                      const float* w1, const float* b1,
                      const float* w2, const float* b2,
                      const float* lnp, float* finalOut)
{
  const int rows = Bc * S;
  const float scale = 0.07216878364870322f;   // 1/sqrt(192)
  for (int il = 0; il < NLc; il++) {
    const float* awl = aw + (size_t)il * 4 * Dc * Dc;
    const float* abl = ab + (size_t)il * 4 * Dc;
    // QKV projections
    launch_gemm(st, X, Dc, 0, awl + 0 * Dc * Dc, Dc, 0, Q,  Dc, 0, abl + 0 * Dc, rows, Dc, Dc, 1.f, 1, false, false);
    launch_gemm(st, X, Dc, 0, awl + 1 * Dc * Dc, Dc, 0, Kb, Dc, 0, abl + 1 * Dc, rows, Dc, Dc, 1.f, 1, false, false);
    launch_gemm(st, X, Dc, 0, awl + 2 * Dc * Dc, Dc, 0, Vb, Dc, 0, abl + 2 * Dc, rows, Dc, Dc, 1.f, 1, false, false);
    // attention, one batch element at a time (bounded logits buffer)
    for (int b = 0; b < Bc; b++) {
      launch_gemm(st, Q + (size_t)b * S * Dc, Dc, (size_t)DHc,
                  Kb + (size_t)b * S * Dc, Dc, (size_t)DHc,
                  LOG, S, (size_t)S * S, nullptr, S, S, DHc, scale, Hc, true, false);
      softmax_kernel<<<dim3(S, Hc), 256, 0, st>>>(LOG, maskf + (size_t)b * S, S);
      launch_gemm(st, LOG, S, (size_t)S * S,
                  Vb + (size_t)b * S * Dc, Dc, (size_t)DHc,
                  Tb + (size_t)b * S * Dc, Dc, (size_t)DHc, nullptr, S, DHc, S, 1.f, Hc, false, false);
    }
    // output projection + LN(x + attn)
    launch_gemm(st, Tb, Dc, 0, awl + 3 * Dc * Dc, Dc, 0, Q, Dc, 0, abl + 3 * Dc, rows, Dc, Dc, 1.f, 1, false, false);
    ln_kernel<<<rows, 128, 0, st>>>(X, Q, lnp + (size_t)il * 4 * Dc, lnp + (size_t)il * 4 * Dc + Dc, X, Dc);
    // FFN (decoder hidden chunked through 24MB scratch)
    int chunk = ((size_t)rows * Fc <= (size_t)Bc * Tc * Dc) ? rows : 4096;
    for (int r0 = 0; r0 < rows; r0 += chunk) {
      launch_gemm(st, X + (size_t)r0 * Dc, Dc, 0, w1 + (size_t)il * Dc * Fc, Fc, 0,
                  Q, Fc, 0, b1 + (size_t)il * Fc, chunk, Fc, Dc, 1.f, 1, false, true);
      launch_gemm(st, Q, Fc, 0, w2 + (size_t)il * Fc * Dc, Dc, 0,
                  Tb + (size_t)r0 * Dc, Dc, 0, b2 + (size_t)il * Dc, chunk, Dc, Fc, 1.f, 1, false, false);
    }
    float* o2 = (il == NLc - 1) ? finalOut : X;
    ln_kernel<<<rows, 128, 0, st>>>(X, Tb, lnp + (size_t)il * 4 * Dc + 2 * Dc,
                                    lnp + (size_t)il * 4 * Dc + 3 * Dc, o2, Dc);
  }
}

extern "C" void kernel_launch(void* const* d_in, const int* in_sizes, int n_in,
                              void* d_out, int out_size, void* d_ws, size_t ws_size,
                              hipStream_t stream)
{
  (void)in_sizes; (void)n_in; (void)out_size; (void)ws_size;
  const int*   ph    = (const int*)d_in[0];
  const int*   dur   = (const int*)d_in[1];
  const float* emb   = (const float*)d_in[2];
  const float* a_enc = (const float*)d_in[3];
  const float* a_dec = (const float*)d_in[4];
  const float* e_aw = (const float*)d_in[5],  *e_ab = (const float*)d_in[6];
  const float* e_w1 = (const float*)d_in[7],  *e_b1 = (const float*)d_in[8];
  const float* e_w2 = (const float*)d_in[9],  *e_b2 = (const float*)d_in[10];
  const float* e_ln = (const float*)d_in[11];
  const float* d_aw = (const float*)d_in[12], *d_ab = (const float*)d_in[13];
  const float* d_w1 = (const float*)d_in[14], *d_b1 = (const float*)d_in[15];
  const float* d_w2 = (const float*)d_in[16], *d_b2 = (const float*)d_in[17];
  const float* d_ln = (const float*)d_in[18];
  const float* c1w  = (const float*)d_in[19], *c1b = (const float*)d_in[20];
  const float* c2w  = (const float*)d_in[21], *c2b = (const float*)d_in[22];
  const float* dpln = (const float*)d_in[23];
  const float* dow  = (const float*)d_in[24], *dob = (const float*)d_in[25];
  const float* melw = (const float*)d_in[26], *melb = (const float*)d_in[27];
  float* out = (float*)d_out;

  // workspace carve-up (~142 MB)
  float* ws = (float*)d_ws;
  size_t off = 0;
  auto alloc = [&](size_t n) { float* p = ws + off; off += n; return p; };
  float* X    = alloc((size_t)Bc * Tc * Dc);    // activations (residual base)
  float* Q    = alloc((size_t)Bc * Tc * Dc);    // q / proj scratch / ffn hidden chunk
  float* Kb   = alloc((size_t)Bc * Tc * Dc);
  float* Vb   = alloc((size_t)Bc * Tc * Dc);
  float* Tb   = alloc((size_t)Bc * Tc * Dc);    // attn concat / ffn out
  float* LOG  = alloc((size_t)Hc * Tc * Tc);    // per-batch logits [H,S,S]
  float* HENC = alloc((size_t)Bc * Lc * Dc);    // encoder output
  float* D1   = alloc((size_t)Bc * Lc * CDPc);
  float* D2   = alloc((size_t)Bc * Lc * CDPc);
  float* ME   = alloc((size_t)Bc * Lc);         // encoder key mask (1 = masked)
  float* MD   = alloc((size_t)Bc * Tc);         // decoder key mask
  int*   CUM  = (int*)alloc((size_t)Bc * Lc);
  int*   IDX  = (int*)alloc((size_t)Bc * Tc);

  // ---- encoder ----
  int nEL = Bc * Lc * Dc;
  embed_pos_kernel<<<(nEL + 255) / 256, 256, 0, stream>>>(ph, emb, a_enc, X, nEL);
  mask_enc_kernel<<<(Bc * Lc + 255) / 256, 256, 0, stream>>>(ph, ME, Bc * Lc);
  run_stack(stream, Lc, X, Q, Kb, Vb, Tb, LOG, ME,
            e_aw, e_ab, e_w1, e_b1, e_w2, e_b2, e_ln, HENC);

  // ---- duration predictor ----
  conv3_relu_kernel<<<dim3(Lc, Bc), CDPc, 0, stream>>>(HENC, c1w, c1b, D1, Lc, Dc, CDPc);
  ln_kernel<<<Bc * Lc, 128, 0, stream>>>(D1, nullptr, dpln + 0, dpln + CDPc, D1, CDPc);
  conv3_relu_kernel<<<dim3(Lc, Bc), CDPc, 0, stream>>>(D1, c2w, c2b, D2, Lc, CDPc, CDPc);
  ln_kernel<<<Bc * Lc, 128, 0, stream>>>(D2, nullptr, dpln + 2 * CDPc, dpln + 3 * CDPc, D2, CDPc);
  dp_dot_kernel<<<Bc * Lc, CDPc, 0, stream>>>(D2, dow, dob, out + (size_t)Bc * Tc * NMELc);

  // ---- length regulator ----
  cumsum_kernel<<<Bc, Lc, 0, stream>>>(dur, CUM);
  lr_idx_kernel<<<(Bc * Tc + 255) / 256, 256, 0, stream>>>(CUM, IDX, MD, Bc * Tc);
  int nDT = Bc * Tc * Dc;
  gather_pos_kernel<<<(nDT + 255) / 256, 256, 0, stream>>>(HENC, IDX, MD, a_dec, X, nDT);

  // ---- decoder ----
  run_stack(stream, Tc, X, Q, Kb, Vb, Tb, LOG, MD,
            d_aw, d_ab, d_w1, d_b1, d_w2, d_b2, d_ln, X);

  // ---- mel projection straight to d_out ----
  launch_gemm(stream, X, Dc, 0, melw, NMELc, 0, out, NMELc, 0, melb,
              Bc * Tc, NMELc, Dc, 1.f, 1, false, false);
}
